// RNN_69818988364524
// MI455X (gfx1250) — matrix operand order repaired and checked by reading, not compiled, not run on hardware
//
#include <hip/hip_runtime.h>
#include <math.h>

#define T_STEPS 512
#define BATCH   128
#define IN_DIM  512
#define LAT     1024
#define NWG     64
#define TPB     256

typedef __bf16 bf16_t;
typedef bf16_t v16bf __attribute__((ext_vector_type(16)));
typedef bf16_t v8bf  __attribute__((ext_vector_type(8)));
typedef float  v8f   __attribute__((ext_vector_type(8)));

// Concatenate two 8-element bf16 chunks into a 16-element WMMA operand.
__device__ __forceinline__ v16bf cat8(v8bf lo, v8bf hi) {
    v16bf r;
#pragma unroll
    for (int j = 0; j < 8; ++j) { r[j] = lo[j]; r[8 + j] = hi[j]; }
    return r;
}

// ---------------------------------------------------------------------------
// Phase 1: xp[t,b,:] = x[t,b,:] @ Wi^T + bi   (M=T*B=65536, N=1024, K=512)
// Block = 256 threads (8 waves). Block tile = 128 rows x 64 cols.
// Wave w owns row-tile w; each wave accumulates 4 column tiles (16x16 each).
// Wi slab (64 cols x 512 K) is converted to bf16 and staged in LDS once.
// ---------------------------------------------------------------------------
__global__ __launch_bounds__(TPB) void xproj_kernel(
    const float* __restrict__ x, const float* __restrict__ Wi,
    const float* __restrict__ bi, float* __restrict__ xp)
{
    __shared__ bf16_t ldsWi[64 * 520];   // row stride 520 bf16 = 1040 B (16B aligned, bank-spread)

    const int tid = threadIdx.x;
    const int c0  = blockIdx.x * 64;

    for (int i = tid; i < 64 * IN_DIM; i += TPB) {
        int r = i >> 9;          // / 512
        int k = i & (IN_DIM - 1);
        ldsWi[r * 520 + k] = (bf16_t)Wi[(size_t)(c0 + r) * IN_DIM + k];
    }
    __syncthreads();

    const int wave = tid >> 5;
    const int lane = tid & 31;
    const int m8   = lane >> 4;   // 0/1 : which K-half this lane holds
    const int lrow = lane & 15;
    const size_t arow = (size_t)blockIdx.y * 128 + wave * 16 + lrow;

    v8f acc[4];
#pragma unroll
    for (int ct = 0; ct < 4; ++ct)
#pragma unroll
        for (int r = 0; r < 8; ++r) acc[ct][r] = 0.0f;

#pragma unroll 2
    for (int kc = 0; kc < IN_DIM / 32; ++kc) {
        const int kb = kc * 32;
        // A fragment: rows of x, K = kb + m8*8 + {0..7} and kb + m8*8 + 16 + {0..7}
        const float4* ap = (const float4*)(x + arow * IN_DIM + kb + m8 * 8);
        float4 f0 = ap[0], f1 = ap[1], f2 = ap[4], f3 = ap[5];
        v16bf a;
        a[0]=(bf16_t)f0.x; a[1]=(bf16_t)f0.y; a[2]=(bf16_t)f0.z; a[3]=(bf16_t)f0.w;
        a[4]=(bf16_t)f1.x; a[5]=(bf16_t)f1.y; a[6]=(bf16_t)f1.z; a[7]=(bf16_t)f1.w;
        a[8]=(bf16_t)f2.x; a[9]=(bf16_t)f2.y; a[10]=(bf16_t)f2.z; a[11]=(bf16_t)f2.w;
        a[12]=(bf16_t)f3.x; a[13]=(bf16_t)f3.y; a[14]=(bf16_t)f3.z; a[15]=(bf16_t)f3.w;

#pragma unroll
        for (int ct = 0; ct < 4; ++ct) {
            // B fragment: col N = ct*16 + lrow, K = kb + m8*8 + {0..7} and kb + m8*8 + 16 + {0..7}
            const v8bf* bp = (const v8bf*)&ldsWi[(ct * 16 + lrow) * 520 + kb + m8 * 8];
            v16bf b = cat8(bp[0], bp[2]);
            acc[ct] = __builtin_amdgcn_wmma_f32_16x16x32_bf16(
                false, a, false, b, (short)0, acc[ct], false, false);
        }
    }

#pragma unroll
    for (int ct = 0; ct < 4; ++ct) {
        const int col = c0 + ct * 16 + lrow;
        const float bv = bi[col];
#pragma unroll
        for (int r = 0; r < 8; ++r) {
            size_t row = (size_t)blockIdx.y * 128 + wave * 16 + m8 * 8 + r;
            xp[row * LAT + col] = acc[ct][r] + bv;
        }
    }
}

// ---------------------------------------------------------------------------
// Phase 1.5: h0 = tanh(xp[0]) -> d_out[0] (f32) and hbuf0 (bf16); init barrier.
// ---------------------------------------------------------------------------
__global__ __launch_bounds__(TPB) void h0_kernel(
    float* __restrict__ out, bf16_t* __restrict__ hbuf0, int* __restrict__ ctr)
{
    int idx = blockIdx.x * TPB + threadIdx.x;
    if (idx == 0) *ctr = 0;
    if (idx < BATCH * LAT) {
        float v = tanhf(out[idx]);
        out[idx]   = v;
        hbuf0[idx] = (bf16_t)v;
    }
}

// ---------------------------------------------------------------------------
// Phase 2: persistent recurrent kernel.
// 64 WGs x 256 threads. WG owns 16 output columns (its Wh slab lives in LDS
// as bf16). Wave w owns rows 16w..16w+15 -> exactly one 16x16 C tile per wave
// per timestep, K=1024 in 32 bf16-WMMA chunks. h ping-pongs between two bf16
// buffers in L2; a global atomic barrier separates timesteps.
// ---------------------------------------------------------------------------
__global__ __launch_bounds__(TPB) void rnn_steps_kernel(
    float* __restrict__ out, const float* __restrict__ Wh,
    const float* __restrict__ bh,
    bf16_t* __restrict__ hbuf0, bf16_t* __restrict__ hbuf1,
    int* ctr)
{
    __shared__ bf16_t ldsWh[16 * 1032];   // row stride 1032 bf16 = 2064 B (bank-spread, 16B aligned)

    const int tid = threadIdx.x;
    const int c0  = blockIdx.x * 16;

    for (int i = tid; i < 16 * LAT; i += TPB) {
        int r = i >> 10;
        int k = i & (LAT - 1);
        ldsWh[r * 1032 + k] = (bf16_t)Wh[(size_t)(c0 + r) * LAT + k];
    }
    __syncthreads();

    const int wave = tid >> 5;
    const int lane = tid & 31;
    const int m8   = lane >> 4;
    const int lrow = lane & 15;
    const int arow = wave * 16 + lrow;     // A row this lane feeds
    const int col  = c0 + lrow;            // C column this lane owns
    const float bhv = bh[col];

    for (int t = 1; t < T_STEPS; ++t) {
        const bf16_t* __restrict__ hin  = (t & 1) ? hbuf0 : hbuf1;
        bf16_t*       __restrict__ hout = (t & 1) ? hbuf1 : hbuf0;

        v8f acc;
#pragma unroll
        for (int r = 0; r < 8; ++r) acc[r] = 0.0f;

#pragma unroll 4
        for (int kc = 0; kc < LAT / 32; ++kc) {
            const int kb = kc * 32;
            const v8bf* ap = (const v8bf*)(hin + (size_t)arow * LAT + kb + m8 * 8);
            v16bf a = cat8(ap[0], ap[2]);          // K = kb+m8*8+{0..7}, kb+m8*8+16+{0..7}
            const v8bf* bp = (const v8bf*)&ldsWh[lrow * 1032 + kb + m8 * 8];
            v16bf b = cat8(bp[0], bp[2]);          // K = kb+m8*8+{0..7}, kb+m8*8+16+{0..7}
            acc = __builtin_amdgcn_wmma_f32_16x16x32_bf16(
                false, a, false, b, (short)0, acc, false, false);
        }

        float* xprow = out + (size_t)t * (BATCH * LAT);
#pragma unroll
        for (int r = 0; r < 8; ++r) {
            int row = wave * 16 + m8 * 8 + r;
            size_t o = (size_t)row * LAT + col;
            float v = tanhf(acc[r] + xprow[o] + bhv);
            xprow[o] = v;                    // f32 output (in place over xp)
            hout[o]  = (bf16_t)v;            // bf16 h for next step's WMMA
        }

        // ---- device-wide barrier between timesteps ----
        __threadfence();
        __syncthreads();
        if (tid == 0) {
            __hip_atomic_fetch_add(ctr, 1, __ATOMIC_ACQ_REL, __HIP_MEMORY_SCOPE_AGENT);
            const int target = NWG * t;
            while (__hip_atomic_load(ctr, __ATOMIC_ACQUIRE, __HIP_MEMORY_SCOPE_AGENT) < target) {
                __builtin_amdgcn_s_sleep(1);
            }
        }
        __syncthreads();
        __threadfence();
    }
}

extern "C" void kernel_launch(void* const* d_in, const int* in_sizes, int n_in,
                              void* d_out, int out_size, void* d_ws, size_t ws_size,
                              hipStream_t stream)
{
    (void)in_sizes; (void)n_in; (void)out_size; (void)ws_size;

    const float* x  = (const float*)d_in[0];
    const float* Wi = (const float*)d_in[1];
    const float* bi = (const float*)d_in[2];
    const float* Wh = (const float*)d_in[3];
    const float* bh = (const float*)d_in[4];
    float* out = (float*)d_out;

    int*    ctr   = (int*)d_ws;
    bf16_t* hbuf0 = (bf16_t*)((char*)d_ws + 256);
    bf16_t* hbuf1 = hbuf0 + (size_t)BATCH * LAT;

    // Phase 1: xp = x @ Wi^T + bi, written into d_out.
    dim3 g1(LAT / 64, (T_STEPS * BATCH) / 128);   // (16, 512)
    xproj_kernel<<<g1, TPB, 0, stream>>>(x, Wi, bi, out);

    // Phase 1.5: h0 = tanh(xp[0]); zero barrier counter.
    h0_kernel<<<(BATCH * LAT) / TPB, TPB, 0, stream>>>(out, hbuf0, ctr);

    // Phase 2: 511 recurrent steps in one persistent kernel.
    rnn_steps_kernel<<<NWG, TPB, 0, stream>>>(out, Wh, bh, hbuf0, hbuf1, ctr);
}
